// GaussianUpsampler_51393578664015
// MI455X (gfx1250) — matrix operand order repaired and checked by reading, not compiled, not run on hardware
//
#include <hip/hip_runtime.h>

#define B_ 32
#define T_ 512
#define D_ 384
#define KSTEPS 16                       // 512 / 32
#define R2PI_INV 0.3989422804014327f    // 1/sqrt(2*pi)

typedef _Float16 v16h __attribute__((ext_vector_type(16)));
typedef _Float16 v8h  __attribute__((ext_vector_type(8)));
typedef float    v8f  __attribute__((ext_vector_type(8)));

#define MTILE 32
#define NTILE 192
#define WSTR  520   // padded LDS stride for weights (halves): 16B-aligned, bank-spread
#define FSTR  40    // padded LDS stride for staged feats tile (halves): 16B-aligned

// ---------------- kernel 1: gaussian centers via per-batch inclusive scan ----
__global__ void gu_centers(const int* __restrict__ dur, float* __restrict__ c_out) {
  __shared__ int s[T_];
  const int b = blockIdx.x;
  const int t = threadIdx.x;
  const int d = dur[b * T_ + t];
  s[t] = d;
  __syncthreads();
  #pragma unroll
  for (int off = 1; off < T_; off <<= 1) {
    int v = (t >= off) ? s[t - off] : 0;
    __syncthreads();
    s[t] += v;
    __syncthreads();
  }
  c_out[b * T_ + t] = (float)s[t] + 0.5f * (float)d;   // c = d/2 + cumsum(d)
}

// ---------------- kernel 1b: feats fp32 -> fp16 in B-fragment-native layout --
// fh[((b*16 + kk)*384 + d)*32 + (t%32)]  : a WMMA B-fragment (16 cols x 32 k)
// is then a contiguous 1KB block -> perfectly coalesced global_load_b128.
__global__ void __launch_bounds__(256)
gu_convert(const float* __restrict__ feats, _Float16* __restrict__ fh) {
  __shared__ _Float16 tile[D_ * 33];
  const int b   = blockIdx.x >> 4;
  const int kk  = blockIdx.x & 15;
  const int tid = threadIdx.x;
  #pragma unroll
  for (int j = 0; j < 48; ++j) {                 // 12288 = 32 t x 384 d
    int idx = j * 256 + tid;
    int tt  = idx / D_;
    int d   = idx - tt * D_;                     // coalesced read along d
    tile[d * 33 + tt] =
        (_Float16)feats[((long long)b * T_ + kk * 32 + tt) * D_ + d];
  }
  __syncthreads();
  _Float16* dst = fh + (long long)(b * KSTEPS + kk) * D_ * 32;
  #pragma unroll
  for (int j = 0; j < 48; ++j) {                 // coalesced write
    int idx = j * 256 + tid;
    dst[idx] = tile[(idx >> 5) * 33 + (idx & 31)];
  }
}

// ---------------- shared device helpers --------------------------------------
__device__ __forceinline__ void gu_weights(const float* __restrict__ rng,
                                           const float* __restrict__ cen,
                                           int b, int o0,
                                           float* c_s, float* invr_s,
                                           _Float16* w_s,
                                           float* rowsum_s, float* invsum_s) {
  const int tid = threadIdx.x;
  #pragma unroll
  for (int i = 0; i < 2; ++i) {
    int t = tid + i * 256;
    invr_s[t] = 1.0f / (rng[b * T_ + t] + 1e-6f);
    c_s[t]    = cen[b * T_ + t];
  }
  if (tid < MTILE) rowsum_s[tid] = 0.0f;
  __syncthreads();
  const int   row = tid >> 3;                    // 8 threads per output row
  const float to  = (float)(o0 + row);
  float sum = 0.0f;
  #pragma unroll 8
  for (int i = 0; i < 64; ++i) {
    int   t  = (tid & 7) + (i << 3);
    float iv = invr_s[t];
    float z  = (to - c_s[t]) * iv;
    float w  = __expf(-0.5f * z * z) * (iv * R2PI_INV) + 1e-6f;
    sum += w;
    w_s[row * WSTR + t] = (_Float16)w;
  }
  atomicAdd(&rowsum_s[row], sum);                // ds_add_f32
  __syncthreads();
  if (tid < MTILE) invsum_s[tid] = 1.0f / rowsum_s[tid];
  __syncthreads();
}

__device__ __forceinline__ void gu_epilogue(float* __restrict__ out,
                                            const float* invsum_s,
                                            int b, int o0, int colbase,
                                            int mb, int g, int outlen,
                                            const v8f& a0, const v8f& a1,
                                            const v8f& a2) {
  #pragma unroll
  for (int v = 0; v < 8; ++v) {
    int row = mb * 16 + g * 8 + v;
    int og  = o0 + row;
    if (og < outlen) {
      float sc = invsum_s[row];
      long long base = ((long long)b * outlen + og) * D_;
      __builtin_nontemporal_store(a0[v] * sc, &out[base + colbase +  0]);
      __builtin_nontemporal_store(a1[v] * sc, &out[base + colbase + 16]);
      __builtin_nontemporal_store(a2[v] * sc, &out[base + colbase + 32]);
    }
  }
}

#define WMMA16(A, Bv, C) \
  __builtin_amdgcn_wmma_f32_16x16x32_f16(false, (A), false, (Bv), (short)0, (C), false, false)

// ---------------- kernel 2a: GEMM, B fragments direct from L2-resident fh ----
__global__ void __launch_bounds__(256)
gu_main_direct(const _Float16* __restrict__ fh, const float* __restrict__ rng,
               const float* __restrict__ cen, float* __restrict__ out, int outlen) {
  __shared__ float    c_s[T_];
  __shared__ float    invr_s[T_];
  __shared__ _Float16 w_s[MTILE * WSTR];
  __shared__ float    rowsum_s[MTILE];
  __shared__ float    invsum_s[MTILE];

  const int tid = threadIdx.x;
  const int b   = blockIdx.z;
  const int n0  = blockIdx.y * NTILE;
  const int o0  = blockIdx.x * MTILE;

  gu_weights(rng, cen, b, o0, c_s, invr_s, w_s, rowsum_s, invsum_s);

  const int lane = tid & 31, wid = tid >> 5;
  const int mb = wid >> 2, nb = wid & 3;
  const int m  = lane & 15, g = lane >> 4;

  v8f acc0 = {}, acc1 = {}, acc2 = {};
  const _Float16* fhb = fh + (long long)b * KSTEPS * D_ * 32
                           + (n0 + nb * 48 + m) * 32 + g * 16;

  for (int kk = 0; kk < KSTEPS; ++kk) {          // no barriers: free pipelining
    const _Float16* pa = &w_s[(mb * 16 + m) * WSTR + kk * 32 + g * 16];
    v8h alo = *(const v8h*)(pa);
    v8h ahi = *(const v8h*)(pa + 8);
    const _Float16* pk = fhb + (long long)kk * D_ * 32;
    v8h b0l = *(const v8h*)(pk +    0), b0h = *(const v8h*)(pk +    8);
    v8h b1l = *(const v8h*)(pk +  512), b1h = *(const v8h*)(pk +  520);
    v8h b2l = *(const v8h*)(pk + 1024), b2h = *(const v8h*)(pk + 1032);
    v16h a  = __builtin_shufflevector(alo, ahi, 0,1,2,3,4,5,6,7,8,9,10,11,12,13,14,15);
    v16h bv0 = __builtin_shufflevector(b0l, b0h, 0,1,2,3,4,5,6,7,8,9,10,11,12,13,14,15);
    v16h bv1 = __builtin_shufflevector(b1l, b1h, 0,1,2,3,4,5,6,7,8,9,10,11,12,13,14,15);
    v16h bv2 = __builtin_shufflevector(b2l, b2h, 0,1,2,3,4,5,6,7,8,9,10,11,12,13,14,15);
    acc0 = WMMA16(a, bv0, acc0);
    acc1 = WMMA16(a, bv1, acc1);
    acc2 = WMMA16(a, bv2, acc2);
  }
  gu_epilogue(out, invsum_s, b, o0, n0 + nb * 48 + m, mb, g, outlen,
              acc0, acc1, acc2);
}

// ---------------- kernel 2b: fallback, LDS-staged feats with reg prefetch ----
__global__ void __launch_bounds__(256)
gu_main_staged(const float* __restrict__ feats, const float* __restrict__ rng,
               const float* __restrict__ cen, float* __restrict__ out, int outlen) {
  __shared__ float    c_s[T_];
  __shared__ float    invr_s[T_];
  __shared__ _Float16 w_s[MTILE * WSTR];
  __shared__ _Float16 fBT[NTILE * FSTR];
  __shared__ float    rowsum_s[MTILE];
  __shared__ float    invsum_s[MTILE];

  const int tid = threadIdx.x;
  const int b   = blockIdx.z;
  const int n0  = blockIdx.y * NTILE;
  const int o0  = blockIdx.x * MTILE;

  gu_weights(rng, cen, b, o0, c_s, invr_s, w_s, rowsum_s, invsum_s);

  const int lane = tid & 31, wid = tid >> 5;
  const int mb = wid >> 2, nb = wid & 3;
  const int m  = lane & 15, g = lane >> 4;

  // thread owns 3 (cc, 8-t-group) strips: coalesced b32 loads, b128 LDS stores
  float pf[24];
  auto load_tile = [&](int kk) {
    #pragma unroll
    for (int j = 0; j < 3; ++j) {
      int idx = j * 256 + tid;                 // 768 strips = 192 cc x 4 groups
      int cc  = idx % NTILE;
      int tt0 = (idx / NTILE) * 8;
      #pragma unroll
      for (int q = 0; q < 8; ++q)
        pf[j * 8 + q] =
            feats[((long long)b * T_ + kk * 32 + tt0 + q) * D_ + n0 + cc];
    }
  };
  load_tile(0);                                 // prefetch first tile

  v8f acc0 = {}, acc1 = {}, acc2 = {};
  for (int kk = 0; kk < KSTEPS; ++kk) {
    __syncthreads();                            // fBT consumed by prev iter
    #pragma unroll
    for (int j = 0; j < 3; ++j) {
      int idx = j * 256 + tid;
      int cc  = idx % NTILE;
      int tt0 = (idx / NTILE) * 8;
      v8h h;
      #pragma unroll
      for (int q = 0; q < 8; ++q) h[q] = (_Float16)pf[j * 8 + q];
      *(v8h*)&fBT[cc * FSTR + tt0] = h;         // ds_store_b128
    }
    __syncthreads();
    if (kk + 1 < KSTEPS) load_tile(kk + 1);     // overlap next loads with wmma

    const _Float16* pa = &w_s[(mb * 16 + m) * WSTR + kk * 32 + g * 16];
    v8h alo = *(const v8h*)(pa);
    v8h ahi = *(const v8h*)(pa + 8);
    const _Float16* pb = &fBT[(nb * 48 + m) * FSTR + g * 16];
    v8h b0l = *(const v8h*)(pb +  0 * FSTR), b0h = *(const v8h*)(pb +  0 * FSTR + 8);
    v8h b1l = *(const v8h*)(pb + 16 * FSTR), b1h = *(const v8h*)(pb + 16 * FSTR + 8);
    v8h b2l = *(const v8h*)(pb + 32 * FSTR), b2h = *(const v8h*)(pb + 32 * FSTR + 8);
    v16h a   = __builtin_shufflevector(alo, ahi, 0,1,2,3,4,5,6,7,8,9,10,11,12,13,14,15);
    v16h bv0 = __builtin_shufflevector(b0l, b0h, 0,1,2,3,4,5,6,7,8,9,10,11,12,13,14,15);
    v16h bv1 = __builtin_shufflevector(b1l, b1h, 0,1,2,3,4,5,6,7,8,9,10,11,12,13,14,15);
    v16h bv2 = __builtin_shufflevector(b2l, b2h, 0,1,2,3,4,5,6,7,8,9,10,11,12,13,14,15);
    acc0 = WMMA16(a, bv0, acc0);
    acc1 = WMMA16(a, bv1, acc1);
    acc2 = WMMA16(a, bv2, acc2);
  }
  gu_epilogue(out, invsum_s, b, o0, n0 + nb * 48 + m, mb, g, outlen,
              acc0, acc1, acc2);
}

extern "C" void kernel_launch(void* const* d_in, const int* in_sizes, int n_in,
                              void* d_out, int out_size, void* d_ws, size_t ws_size,
                              hipStream_t stream) {
  const float* feats = (const float*)d_in[0];
  const float* rng   = (const float*)d_in[1];
  const int*   dur   = (const int*)d_in[2];
  float* out = (float*)d_out;
  const int outlen = out_size / (B_ * D_);           // out is [B, outlen, D]

  const size_t CEN_BYTES = (size_t)B_ * T_ * sizeof(float);
  const size_t FH_BYTES  = (size_t)B_ * T_ * D_ * sizeof(_Float16);
  float* cen = (float*)d_ws;

  gu_centers<<<dim3(B_), dim3(T_), 0, stream>>>(dur, cen);

  dim3 grid((outlen + MTILE - 1) / MTILE, D_ / NTILE, B_);
  if (ws_size >= CEN_BYTES + FH_BYTES) {
    _Float16* fh = (_Float16*)((char*)d_ws + CEN_BYTES);
    gu_convert<<<dim3(B_ * KSTEPS), dim3(256), 0, stream>>>(feats, fh);
    gu_main_direct<<<grid, dim3(256), 0, stream>>>(fh, rng, cen, out, outlen);
  } else {
    gu_main_staged<<<grid, dim3(256), 0, stream>>>(feats, rng, cen, out, outlen);
  }
}
